// Seq2Seq_35192962023621
// MI455X (gfx1250) — matrix operand order repaired and checked by reading, not compiled, not run on hardware
//
#include <hip/hip_runtime.h>
#include <hip/hip_bf16.h>
#include <math.h>

typedef __attribute__((ext_vector_type(16))) _Float16 v16h;
typedef __attribute__((ext_vector_type(8)))  _Float16 v8h;
typedef __attribute__((ext_vector_type(8)))  float    v8f;

#define BB 32
#define SS 64
#define TT 48
#define EE 256
#define EH 512
#define DH 512
#define VV 32000

// ---------------------------------------------------------------------------
// Generic WMMA GEMM: C[M,N] = A[M,K](f16) * W[N,K]^T(f16) + bias, optional tanh,
// optional f16 mirror Ch, and a (t,b) row remap for the final output GEMM.
// Each wave computes a 32x64 strip of C (2 M-tiles x 4 N-tiles) so every
// loaded B fragment feeds two v_wmma_f32_16x16x32_f16 ops -> halves L2 B traffic.
// A layout per ISA: lane<16 -> M=lane, K=[0..7 | 16..23]; lane>=16 -> K=[8..15 | 24..31].
// B layout per ISA: lane<16 -> N=lane, K=[0..7 | 16..23]; lane>=16 -> K=[8..15 | 24..31].
// M must be a multiple of 32, N a multiple of 64, K a multiple of 32.
// ---------------------------------------------------------------------------
__global__ __launch_bounds__(256) void gemm_wmma(
    const _Float16* __restrict__ A, const _Float16* __restrict__ Wt,
    const float* __restrict__ bias, float* __restrict__ C, _Float16* __restrict__ Ch,
    int M, int N, int K, int act, int rowdiv, long long strideHi, long long strideLo)
{
  int lane = threadIdx.x & 31;
  int wave = threadIdx.x >> 5;
  int nstrips = N >> 6;                 // N / 64
  int jobs = (M >> 5) * nstrips;        // 32-row jobs
  int job = blockIdx.x * 8 + wave;
  if (job >= jobs) return;
  int mpair  = job / nstrips;
  int nstrip = job % nstrips;
  int nbase  = nstrip << 6;
  int mbase32 = mpair << 5;

  int aoff = (lane & 16) ? 8 : 0;       // K sub-offset for A halves
  int boff = (lane & 16) ? 8 : 0;       // K sub-offset for B halves
  int nn = lane & 15;

  const _Float16* Arow0 = A + (size_t)(mbase32 + (lane & 15)) * K;
  const _Float16* Arow1 = Arow0 + (size_t)16 * K;

  const _Float16* Brow0 = Wt + (size_t)(nbase + 0 * 16 + nn) * K + boff;
  const _Float16* Brow1 = Wt + (size_t)(nbase + 1 * 16 + nn) * K + boff;
  const _Float16* Brow2 = Wt + (size_t)(nbase + 2 * 16 + nn) * K + boff;
  const _Float16* Brow3 = Wt + (size_t)(nbase + 3 * 16 + nn) * K + boff;

  v8f acc[8];
#pragma unroll
  for (int i = 0; i < 8; i++) acc[i] = (v8f){};

  for (int kb = 0; kb < K; kb += 32) {
    v8h a0l = *(const v8h*)(Arow0 + kb + aoff);
    v8h a0h = *(const v8h*)(Arow0 + kb + aoff + 16);
    v8h a1l = *(const v8h*)(Arow1 + kb + aoff);
    v8h a1h = *(const v8h*)(Arow1 + kb + aoff + 16);
    v16h a0, a1;
#pragma unroll
    for (int i = 0; i < 8; i++) {
      a0[i] = a0l[i]; a0[i + 8] = a0h[i];
      a1[i] = a1l[i]; a1[i + 8] = a1h[i];
    }

    v8h b0l = *(const v8h*)(Brow0 + kb);
    v8h b0h = *(const v8h*)(Brow0 + kb + 16);
    v8h b1l = *(const v8h*)(Brow1 + kb);
    v8h b1h = *(const v8h*)(Brow1 + kb + 16);
    v8h b2l = *(const v8h*)(Brow2 + kb);
    v8h b2h = *(const v8h*)(Brow2 + kb + 16);
    v8h b3l = *(const v8h*)(Brow3 + kb);
    v8h b3h = *(const v8h*)(Brow3 + kb + 16);
    v16h b0, b1, b2, b3;
#pragma unroll
    for (int i = 0; i < 8; i++) {
      b0[i] = b0l[i]; b0[i + 8] = b0h[i];
      b1[i] = b1l[i]; b1[i + 8] = b1h[i];
      b2[i] = b2l[i]; b2[i + 8] = b2h[i];
      b3[i] = b3l[i]; b3[i + 8] = b3h[i];
    }
    // each B fragment used by both M-tiles (B reuse in registers)
    acc[0] = __builtin_amdgcn_wmma_f32_16x16x32_f16(false, a0, false, b0, (short)0, acc[0], false, false);
    acc[4] = __builtin_amdgcn_wmma_f32_16x16x32_f16(false, a1, false, b0, (short)0, acc[4], false, false);
    acc[1] = __builtin_amdgcn_wmma_f32_16x16x32_f16(false, a0, false, b1, (short)0, acc[1], false, false);
    acc[5] = __builtin_amdgcn_wmma_f32_16x16x32_f16(false, a1, false, b1, (short)0, acc[5], false, false);
    acc[2] = __builtin_amdgcn_wmma_f32_16x16x32_f16(false, a0, false, b2, (short)0, acc[2], false, false);
    acc[6] = __builtin_amdgcn_wmma_f32_16x16x32_f16(false, a1, false, b2, (short)0, acc[6], false, false);
    acc[3] = __builtin_amdgcn_wmma_f32_16x16x32_f16(false, a0, false, b3, (short)0, acc[3], false, false);
    acc[7] = __builtin_amdgcn_wmma_f32_16x16x32_f16(false, a1, false, b3, (short)0, acc[7], false, false);
  }

  // Store: C/D layout: VGPR v -> M = tilebase + v + (lane<16 ? 0 : 8); lane&15 -> N
  int mlo = mbase32 + ((lane & 16) ? 8 : 0);
#pragma unroll
  for (int half = 0; half < 2; half++) {
    int mb = mlo + half * 16;
#pragma unroll
    for (int s = 0; s < 4; s++) {
      int ncol = nbase + s * 16 + nn;
      float bv = bias ? bias[ncol] : 0.0f;
      v8f av = acc[half * 4 + s];
#pragma unroll
      for (int v = 0; v < 8; v++) {
        int m = mb + v;
        float val = av[v] + bv;
        if (act == 1) val = tanhf(val);
        size_t off = (size_t)(m / rowdiv) * (size_t)strideHi +
                     (size_t)(m % rowdiv) * (size_t)strideLo + (size_t)ncol;
        C[off] = val;
        if (Ch) Ch[(size_t)m * N + ncol] = (_Float16)val;
      }
    }
  }
}

// f32 -> f16 with optional row stride/offset (used for Wq/We extraction too)
__global__ void cvt_f16(const float* __restrict__ src, _Float16* __restrict__ dst,
                        long long total, int cols, int srcStride, int srcOff)
{
  for (long long i = (long long)blockIdx.x * blockDim.x + threadIdx.x; i < total;
       i += (long long)gridDim.x * blockDim.x) {
    long long r = i / cols; int c = (int)(i % cols);
    dst[i] = (_Float16)src[r * (long long)srcStride + srcOff + c];
  }
}

// gather encoder embeddings: x16[(s*B+b), :] = enc_emb[src[b,s], :]
__global__ void embed_src(const int* __restrict__ src, const float* __restrict__ emb,
                          _Float16* __restrict__ x16)
{
  int row = blockIdx.x;              // row = s*B + b
  int s = row / BB, b = row % BB;
  int tok = src[b * SS + s];
  x16[(size_t)row * EE + threadIdx.x] = (_Float16)emb[(size_t)tok * EE + threadIdx.x];
}

// gather decoder embeddings into rnn input buffer [emb|w] and Z [h2|w|emb]
__global__ void embed_trg(const int* __restrict__ trg, const float* __restrict__ emb,
                          _Float16* __restrict__ rnnbuf, _Float16* __restrict__ Z)
{
  int row = blockIdx.x;              // row = t*B + b
  int t = row / BB, b = row % BB;
  int tok = trg[b * TT + t];
  float v = emb[(size_t)tok * EE + threadIdx.x];
  rnnbuf[(size_t)row * (EE + 2 * EH) + threadIdx.x] = (_Float16)v;
  Z[(size_t)row * (DH + 2 * EH + EE) + (DH + 2 * EH) + threadIdx.x] = (_Float16)v;
}

// GRU elementwise update; optional f32/f16 side outputs (enc_out slot, Z slot)
__global__ void gru_elem(const float* __restrict__ gi, const float* __restrict__ gh,
                         float* __restrict__ h, _Float16* __restrict__ h16,
                         float* __restrict__ aux, long long auxStride,
                         _Float16* __restrict__ aux2, long long aux2Stride, int H)
{
  int i = blockIdx.x * blockDim.x + threadIdx.x;
  if (i >= BB * H) return;
  int b = i / H, j = i % H;
  const float* gib = gi + (size_t)b * 3 * H;
  const float* ghb = gh + (size_t)b * 3 * H;
  float r = 1.0f / (1.0f + expf(-(gib[j] + ghb[j])));
  float z = 1.0f / (1.0f + expf(-(gib[H + j] + ghb[H + j])));
  float n = tanhf(gib[2 * H + j] + r * ghb[2 * H + j]);
  float h2 = (1.0f - z) * n + z * h[i];
  h[i] = h2;
  if (h16) h16[i] = (_Float16)h2;
  if (aux) aux[(size_t)b * auxStride + j] = h2;
  if (aux2) aux2[(size_t)b * aux2Stride + j] = (_Float16)h2;
}

__global__ void hcat_kernel(const float* __restrict__ hf, const float* __restrict__ hb,
                            _Float16* __restrict__ out)
{
  int i = blockIdx.x * 256 + threadIdx.x;
  if (i >= BB * 2 * EH) return;
  int b = i >> 10, j = i & 1023;
  float v = (j < EH) ? hf[b * EH + j] : hb[b * EH + (j - EH)];
  out[i] = (_Float16)v;
}

// attention for decoder step t: scores = v . tanh(qW + encAtt), softmax over s,
// w = sum a * enc_out; writes f16 w into rnnbuf[t] and Z[t]
__global__ void attn_kernel(const float* __restrict__ qW, const float* __restrict__ encAtt,
                            const float* __restrict__ v_attn, const float* __restrict__ enc_out,
                            _Float16* __restrict__ rnn_t, _Float16* __restrict__ Z_t)
{
  int b = blockIdx.x, tid = threadIdx.x;
  __shared__ float red[256];
  __shared__ float sc[SS];
  int s = tid >> 2, part = tid & 3;
  const float* ea = encAtt + ((size_t)b * SS + s) * DH;
  const float* q  = qW + (size_t)b * DH;
  float p = 0.0f;
  for (int d = part; d < DH; d += 4) p += v_attn[d] * tanhf(q[d] + ea[d]);
  red[tid] = p;
  __syncthreads();
  if (part == 0) sc[s] = red[tid] + red[tid + 1] + red[tid + 2] + red[tid + 3];
  __syncthreads();
  if (tid == 0) {
    float mx = sc[0];
    for (int i = 1; i < SS; i++) mx = fmaxf(mx, sc[i]);
    float sum = 0.0f;
    for (int i = 0; i < SS; i++) { sc[i] = expf(sc[i] - mx); sum += sc[i]; }
    float inv = 1.0f / sum;
    for (int i = 0; i < SS; i++) sc[i] *= inv;
  }
  __syncthreads();
  for (int d = tid; d < 2 * EH; d += 256) {
    float acc = 0.0f;
    const float* eo = enc_out + (size_t)b * SS * (2 * EH) + d;
    for (int i = 0; i < SS; i++) acc += sc[i] * eo[(size_t)i * (2 * EH)];
    rnn_t[(size_t)b * (EE + 2 * EH) + EE + d] = (_Float16)acc;
    Z_t[(size_t)b * (DH + 2 * EH + EE) + DH + d] = (_Float16)acc;
  }
}

// ---------------------------------------------------------------------------

static inline void launch_gemm(hipStream_t st, const _Float16* A, const _Float16* Wt,
                               const float* bias, float* C, _Float16* Ch,
                               int M, int N, int K, int act,
                               int rowdiv, long long sH, long long sL)
{
  int jobs = (M / 32) * (N / 64);
  int blocks = (jobs + 7) / 8;
  hipLaunchKernelGGL(gemm_wmma, dim3(blocks), dim3(256), 0, st,
                     A, Wt, bias, C, Ch, M, N, K, act, rowdiv, sH, sL);
}

extern "C" void kernel_launch(void* const* d_in, const int* in_sizes, int n_in,
                              void* d_out, int out_size, void* d_ws, size_t ws_size,
                              hipStream_t stream)
{
  const int*   src     = (const int*)  d_in[0];
  const int*   trg     = (const int*)  d_in[1];
  const float* enc_emb = (const float*)d_in[2];
  const float* W_ih_f  = (const float*)d_in[3];
  const float* W_hh_f  = (const float*)d_in[4];
  const float* b_ih_f  = (const float*)d_in[5];
  const float* b_hh_f  = (const float*)d_in[6];
  const float* W_ih_b  = (const float*)d_in[7];
  const float* W_hh_b  = (const float*)d_in[8];
  const float* b_ih_b  = (const float*)d_in[9];
  const float* b_hh_b  = (const float*)d_in[10];
  const float* W_fc    = (const float*)d_in[11];
  const float* b_fc    = (const float*)d_in[12];
  const float* W_attn  = (const float*)d_in[13];
  const float* b_attn  = (const float*)d_in[14];
  const float* v_attn  = (const float*)d_in[15];
  const float* dec_emb = (const float*)d_in[16];
  const float* W_ih_d  = (const float*)d_in[17];
  const float* W_hh_d  = (const float*)d_in[18];
  const float* b_ih_d  = (const float*)d_in[19];
  const float* b_hh_d  = (const float*)d_in[20];
  const float* W_out   = (const float*)d_in[21];
  const float* b_out   = (const float*)d_in[22];
  float* out = (float*)d_out;

  // ---- workspace carve (256B aligned) ----
  char* p = (char*)d_ws;
  auto alloc = [&](size_t bytes) -> char* {
    char* r = p; p += (bytes + 255) & ~(size_t)255; return r;
  };
  const int K_rnn = EE + 2 * EH;            // 1280
  const int K_out = DH + 2 * EH + EE;       // 1792

  _Float16* W_ihf16  = (_Float16*)alloc((size_t)3 * EH * EE * 2);
  _Float16* W_hhf16  = (_Float16*)alloc((size_t)3 * EH * EH * 2);
  _Float16* W_ihb16  = (_Float16*)alloc((size_t)3 * EH * EE * 2);
  _Float16* W_hhb16  = (_Float16*)alloc((size_t)3 * EH * EH * 2);
  _Float16* W_fc16   = (_Float16*)alloc((size_t)DH * 2 * EH * 2);
  _Float16* Wq16     = (_Float16*)alloc((size_t)DH * DH * 2);
  _Float16* We16     = (_Float16*)alloc((size_t)DH * 2 * EH * 2);
  _Float16* W_ihd16  = (_Float16*)alloc((size_t)3 * DH * K_rnn * 2);
  _Float16* W_hhd16  = (_Float16*)alloc((size_t)3 * DH * DH * 2);
  _Float16* W_out16  = (_Float16*)alloc((size_t)VV * K_out * 2);
  _Float16* x16      = (_Float16*)alloc((size_t)SS * BB * EE * 2);
  _Float16* encout16 = (_Float16*)alloc((size_t)BB * SS * 2 * EH * 2);
  _Float16* rnnbuf   = (_Float16*)alloc((size_t)TT * BB * K_rnn * 2);
  _Float16* Z        = (_Float16*)alloc((size_t)TT * BB * K_out * 2);
  _Float16* hf16     = (_Float16*)alloc((size_t)BB * EH * 2);
  _Float16* hb16     = (_Float16*)alloc((size_t)BB * EH * 2);
  _Float16* hdec16   = (_Float16*)alloc((size_t)BB * DH * 2);
  _Float16* hcat16   = (_Float16*)alloc((size_t)BB * 2 * EH * 2);

  float* gi_f   = (float*)alloc((size_t)SS * BB * 3 * EH * 4);
  float* gi_b   = (float*)alloc((size_t)SS * BB * 3 * EH * 4);
  float* gh_f   = (float*)alloc((size_t)BB * 3 * EH * 4);
  float* gh_b   = (float*)alloc((size_t)BB * 3 * EH * 4);
  float* encout = (float*)alloc((size_t)BB * SS * 2 * EH * 4);
  float* encAtt = (float*)alloc((size_t)BB * SS * DH * 4);
  float* h_f    = (float*)alloc((size_t)BB * EH * 4);
  float* h_b    = (float*)alloc((size_t)BB * EH * 4);
  float* h_dec  = (float*)alloc((size_t)BB * DH * 4);
  float* qW     = (float*)alloc((size_t)BB * DH * 4);
  float* gi_d   = (float*)alloc((size_t)BB * 3 * DH * 4);
  float* gh_d   = (float*)alloc((size_t)BB * 3 * DH * 4);
  (void)ws_size; (void)n_in; (void)in_sizes; (void)out_size;

  // ---- weight conversions to f16 ----
  auto cvt = [&](const float* s, _Float16* d, long long total, int cols, int stride, int off) {
    int blocks = (int)((total + 255) / 256); if (blocks > 4096) blocks = 4096;
    hipLaunchKernelGGL(cvt_f16, dim3(blocks), dim3(256), 0, stream, s, d, total, cols, stride, off);
  };
  cvt(W_ih_f, W_ihf16, (long long)3 * EH * EE, EE, EE, 0);
  cvt(W_hh_f, W_hhf16, (long long)3 * EH * EH, EH, EH, 0);
  cvt(W_ih_b, W_ihb16, (long long)3 * EH * EE, EE, EE, 0);
  cvt(W_hh_b, W_hhb16, (long long)3 * EH * EH, EH, EH, 0);
  cvt(W_fc,   W_fc16,  (long long)DH * 2 * EH, 2 * EH, 2 * EH, 0);
  cvt(W_attn, Wq16,    (long long)DH * DH, DH, DH + 2 * EH, 0);          // cols 0..511
  cvt(W_attn, We16,    (long long)DH * 2 * EH, 2 * EH, DH + 2 * EH, DH); // cols 512..1535
  cvt(W_ih_d, W_ihd16, (long long)3 * DH * K_rnn, K_rnn, K_rnn, 0);
  cvt(W_hh_d, W_hhd16, (long long)3 * DH * DH, DH, DH, 0);
  cvt(W_out,  W_out16, (long long)VV * K_out, K_out, K_out, 0);

  // ---- embeddings ----
  hipLaunchKernelGGL(embed_src, dim3(SS * BB), dim3(EE), 0, stream, src, enc_emb, x16);
  hipLaunchKernelGGL(embed_trg, dim3(TT * BB), dim3(EE), 0, stream, trg, dec_emb, rnnbuf, Z);

  // ---- zero-init hidden states ----
  hipMemsetAsync(h_f, 0, (size_t)BB * EH * 4, stream);
  hipMemsetAsync(h_b, 0, (size_t)BB * EH * 4, stream);
  hipMemsetAsync(hf16, 0, (size_t)BB * EH * 2, stream);
  hipMemsetAsync(hb16, 0, (size_t)BB * EH * 2, stream);

  // ---- encoder input-side GEMMs (hoisted over time): gi = x @ W_ih^T + b_ih ----
  launch_gemm(stream, x16, W_ihf16, b_ih_f, gi_f, nullptr, SS * BB, 3 * EH, EE, 0, 1, 3 * EH, 0);
  launch_gemm(stream, x16, W_ihb16, b_ih_b, gi_b, nullptr, SS * BB, 3 * EH, EE, 0, 1, 3 * EH, 0);

  // ---- recurrent encoder: 64 sequential steps, fwd + bwd ----
  int gruBlocks = (BB * EH + 255) / 256;
  for (int i = 0; i < SS; i++) {
    int sF = i, sB = SS - 1 - i;
    // forward dir
    launch_gemm(stream, hf16, W_hhf16, b_hh_f, gh_f, nullptr, BB, 3 * EH, EH, 0, 1, 3 * EH, 0);
    hipLaunchKernelGGL(gru_elem, dim3(gruBlocks), dim3(256), 0, stream,
                       gi_f + (size_t)sF * BB * 3 * EH, gh_f, h_f, hf16,
                       encout + (size_t)sF * 2 * EH + 0, (long long)SS * 2 * EH,
                       (_Float16*)nullptr, 0LL, EH);
    // backward dir
    launch_gemm(stream, hb16, W_hhb16, b_hh_b, gh_b, nullptr, BB, 3 * EH, EH, 0, 1, 3 * EH, 0);
    hipLaunchKernelGGL(gru_elem, dim3(gruBlocks), dim3(256), 0, stream,
                       gi_b + (size_t)sB * BB * 3 * EH, gh_b, h_b, hb16,
                       encout + (size_t)sB * 2 * EH + EH, (long long)SS * 2 * EH,
                       (_Float16*)nullptr, 0LL, EH);
  }

  // ---- decoder init: hidden = tanh([hF|hB] @ W_fc^T + b_fc) ----
  hipLaunchKernelGGL(hcat_kernel, dim3((BB * 2 * EH + 255) / 256), dim3(256), 0, stream,
                     h_f, h_b, hcat16);
  launch_gemm(stream, hcat16, W_fc16, b_fc, h_dec, hdec16, BB, DH, 2 * EH, 1, 1, DH, 0);

  // ---- hoisted attention precompute: encAtt = enc_out @ We^T + b_attn ----
  cvt(encout, encout16, (long long)BB * SS * 2 * EH, 2 * EH, 2 * EH, 0);
  launch_gemm(stream, encout16, We16, b_attn, encAtt, nullptr, BB * SS, DH, 2 * EH, 0, 1, DH, 0);

  // ---- decoder: 48 sequential steps ----
  int gruBlocksD = (BB * DH + 255) / 256;
  for (int t = 0; t < TT; t++) {
    _Float16* rnn_t = rnnbuf + (size_t)t * BB * K_rnn;
    _Float16* Z_t   = Z + (size_t)t * BB * K_out;
    // qW = h @ Wq^T
    launch_gemm(stream, hdec16, Wq16, nullptr, qW, nullptr, BB, DH, DH, 0, 1, DH, 0);
    // attention: softmax(v . tanh(qW + encAtt)), context w -> rnnbuf[t], Z[t]
    hipLaunchKernelGGL(attn_kernel, dim3(BB), dim3(256), 0, stream,
                       qW, encAtt, v_attn, encout, rnn_t, Z_t);
    // gi = [emb|w] @ W_ih_d^T + b ; gh = h @ W_hh_d^T + b
    launch_gemm(stream, rnn_t, W_ihd16, b_ih_d, gi_d, nullptr, BB, 3 * DH, K_rnn, 0, 1, 3 * DH, 0);
    launch_gemm(stream, hdec16, W_hhd16, b_hh_d, gh_d, nullptr, BB, 3 * DH, DH, 0, 1, 3 * DH, 0);
    // h2 -> h_dec (f32+f16) and Z[t][:,0:512] (f16)
    hipLaunchKernelGGL(gru_elem, dim3(gruBlocksD), dim3(256), 0, stream,
                       gi_d, gh_d, h_dec, hdec16,
                       (float*)nullptr, 0LL, Z_t, (long long)K_out, DH);
  }

  // ---- batched output projection: preds = Z @ W_out^T + b_out, remap (t,b)->out[b,t,:] ----
  launch_gemm(stream, Z, W_out16, b_out, out, nullptr, TT * BB, VV, K_out, 0,
              BB, (long long)VV, (long long)TT * VV);
}
